// DescriptorMatcher_8272107012371
// MI455X (gfx1250) — compile-verified, matrix operand order checked by reading, not run on hardware
//
#include <hip/hip_runtime.h>

// ---------------------------------------------------------------------------
// DescriptorMatcher NN (8192x8192, D=128) on MI455X (gfx1250, wave32).
// dist^2 = |a|^2 + |b|^2 - 2 a.b ; argmin over b.  GEMM via
// v_wmma_f32_16x16x32_bf16 with a bf16 hi/lo split (hi*hi + lo*hi + hi*lo,
// f32 accumulate) for ~fp32 accuracy at bf16 matrix-core rate.
// Argmin fused; B tiles register-double-buffered to overlap VMEM with WMMA.
// ---------------------------------------------------------------------------

#define B1N 8192
#define B2N 8192
#define DD  128
#define NSPLIT 8
#define MROWS_PER_WG 128   // 8 waves * 16 rows

typedef __attribute__((ext_vector_type(16))) __bf16 bf16x16;
typedef __attribute__((ext_vector_type(8)))  __bf16 bf16x8;
typedef __attribute__((ext_vector_type(8)))  float  f32x8;

static __device__ __forceinline__ bf16x16 load_b16x16(const __bf16* p) {
  bf16x8 a = *(const bf16x8*)(p);
  bf16x8 b = *(const bf16x8*)(p + 16);
  return __builtin_shufflevector(a, b, 0, 1, 2, 3, 4, 5, 6, 7,
                                       8, 9, 10, 11, 12, 13, 14, 15);
}

// ---------------------------------------------------------------------------
// Prep: split f32 rows into bf16 hi (truncated, exact) + bf16 lo (residual,
// RNE) and compute full-precision row squared norms.  One block per row.
// ---------------------------------------------------------------------------
__global__ __launch_bounds__(128) void prep_kernel(const float* __restrict__ src,
                                                   __bf16* __restrict__ hi,
                                                   __bf16* __restrict__ lo,
                                                   float* __restrict__ nrm) {
  const int row = blockIdx.x;
  const int k   = threadIdx.x;
  const int idx = row * DD + k;

  float x = src[idx];
  union { float f; unsigned u; } c;
  c.f = x;
  c.u &= 0xFFFF0000u;            // exact bf16 prefix
  float h = c.f;
  float r = x - h;               // exact residual in f32
  hi[idx] = (__bf16)h;
  lo[idx] = (__bf16)r;

  float s = x * x;
#pragma unroll
  for (int m = 1; m < 32; m <<= 1) s += __shfl_xor(s, m, 32);

  __shared__ float sm[4];
  const int wave = threadIdx.x >> 5;
  if ((threadIdx.x & 31) == 0) sm[wave] = s;
  __syncthreads();
  if (threadIdx.x == 0) nrm[row] = sm[0] + sm[1] + sm[2] + sm[3];
}

// ---------------------------------------------------------------------------
// B tile held in registers: 4 K-chunks x (hi, lo) + column norm + column idx.
// ---------------------------------------------------------------------------
struct TileB {
  bf16x16 h[4];
  bf16x16 l[4];
  float   nbv;
  int     bRow;
};

static __device__ __forceinline__ void load_tileB(TileB& tb,
                                                  const __bf16* __restrict__ bHi,
                                                  const __bf16* __restrict__ bLo,
                                                  const float* __restrict__ nb,
                                                  int n0, int l15, int koff) {
  tb.bRow = n0 + l15;
  const __bf16* pb = bHi + (size_t)tb.bRow * DD + koff;
  const __bf16* qb = bLo + (size_t)tb.bRow * DD + koff;
#pragma unroll
  for (int kc = 0; kc < 4; ++kc) {
    tb.h[kc] = load_b16x16(pb + kc * 32);
    tb.l[kc] = load_b16x16(qb + kc * 32);
  }
  tb.nbv = nb[tb.bRow];
}

static __device__ __forceinline__ void compute_tile(const TileB& tb,
                                                    const bf16x16 ah[4],
                                                    const bf16x16 al[4],
                                                    float bestS[8], int bestI[8]) {
  // Three independent accumulation chains (depth 4 each) so the scheduler can
  // interleave WMMAs instead of one 12-deep RAW chain.
  f32x8 acc0 = {}, acc1 = {}, acc2 = {};
#pragma unroll
  for (int kc = 0; kc < 4; ++kc) {
    acc0 = __builtin_amdgcn_wmma_f32_16x16x32_bf16(
        false, ah[kc], false, tb.h[kc], (short)0, acc0, false, false);
    acc1 = __builtin_amdgcn_wmma_f32_16x16x32_bf16(
        false, al[kc], false, tb.h[kc], (short)0, acc1, false, false);
    acc2 = __builtin_amdgcn_wmma_f32_16x16x32_bf16(
        false, ah[kc], false, tb.l[kc], (short)0, acc2, false, false);
  }
#pragma unroll
  for (int v = 0; v < 8; ++v) {
    float dot = acc0[v] + acc1[v] + acc2[v];
    float sc  = __builtin_fmaf(-2.0f, dot, tb.nbv);
    if (sc < bestS[v]) { bestS[v] = sc; bestI[v] = tb.bRow; }
  }
}

// ---------------------------------------------------------------------------
// Fused WMMA GEMM + per-N-split argmin.
// grid = (NSPLIT, B1/128); block = 256 threads = 8 waves.
// Wave w owns rows [blockIdx.y*128 + w*16, +16); ping-pong double-buffers the
// streamed B tiles in registers so loads for tile t+1 are in flight while the
// WMMAs for tile t execute.
// ---------------------------------------------------------------------------
__global__ __launch_bounds__(256) void nn_wmma_kernel(
    const __bf16* __restrict__ aHi, const __bf16* __restrict__ aLo,
    const __bf16* __restrict__ bHi, const __bf16* __restrict__ bLo,
    const float*  __restrict__ nb,
    float* __restrict__ pS, int* __restrict__ pI) {
  const int lane = threadIdx.x & 31;
  const int wave = threadIdx.x >> 5;
  const int l15  = lane & 15;
  const int koff = (lane >> 4) << 3;   // 0 for lanes 0-15, 8 for lanes 16-31
  const int mRow = blockIdx.y * MROWS_PER_WG + wave * 16 + l15;
  const int nsplit = blockIdx.x;
  const int nPerSplit = B2N / NSPLIT;  // 1024 columns per split
  const int NT = nPerSplit / 16;       // 64 tiles

  // Resident A operands: 4 K-chunks x (hi, lo).
  bf16x16 ah[4], al[4];
#pragma unroll
  for (int kc = 0; kc < 4; ++kc) {
    ah[kc] = load_b16x16(aHi + (size_t)mRow * DD + kc * 32 + koff);
    al[kc] = load_b16x16(aLo + (size_t)mRow * DD + kc * 32 + koff);
  }

  float bestS[8];
  int   bestI[8];
#pragma unroll
  for (int v = 0; v < 8; ++v) { bestS[v] = __builtin_inff(); bestI[v] = 0; }

  const int n0base = nsplit * nPerSplit;

  TileB tA, tB;
  load_tileB(tA, bHi, bLo, nb, n0base, l15, koff);

  for (int t = 0; t < NT; t += 2) {
    // Issue loads for tile t+1, then prefetch further ahead into near cache.
    load_tileB(tB, bHi, bLo, nb, n0base + (t + 1) * 16, l15, koff);
    if (t + 4 < NT) {
      const size_t pfRow = (size_t)(n0base + (t + 4) * 16 + l15) * DD;
      __builtin_prefetch(bHi + pfRow, 0, 3);
      __builtin_prefetch(bLo + pfRow, 0, 3);
    }
    compute_tile(tA, ah, al, bestS, bestI);

    if (t + 2 < NT) load_tileB(tA, bHi, bLo, nb, n0base + (t + 2) * 16, l15, koff);
    compute_tile(tB, ah, al, bestS, bestI);
  }

  // Min-reduce across the 16 lanes of each half (ties -> smaller index).
#pragma unroll
  for (int v = 0; v < 8; ++v) {
    float s = bestS[v];
    int   i = bestI[v];
#pragma unroll
    for (int m = 1; m < 16; m <<= 1) {
      float s2 = __shfl_xor(s, m, 32);
      int   i2 = __shfl_xor(i, m, 32);
      if (s2 < s || (s2 == s && i2 < i)) { s = s2; i = i2; }
    }
    bestS[v] = s;
    bestI[v] = i;
  }

  if (l15 == 0) {
    // acc slot v holds row M = v + 8*(lane>=16) of this wave's 16-row strip.
    const int rbase = blockIdx.y * MROWS_PER_WG + wave * 16 + koff;
#pragma unroll
    for (int v = 0; v < 8; ++v) {
      pS[(size_t)(rbase + v) * NSPLIT + nsplit] = bestS[v];
      pI[(size_t)(rbase + v) * NSPLIT + nsplit] = bestI[v];
    }
  }
}

// ---------------------------------------------------------------------------
// Finalize: reduce NSPLIT partial winners per row, recover the true distance
// dist = sqrt(clip(|a|^2 + score, 0)), emit dists (f32) and [i, j] (int32).
// ---------------------------------------------------------------------------
__global__ __launch_bounds__(256) void finalize_kernel(
    const float* __restrict__ pS, const int* __restrict__ pI,
    const float* __restrict__ na,
    float* __restrict__ outD, int* __restrict__ outIdx) {
  const int row = blockIdx.x * blockDim.x + threadIdx.x;
  if (row >= B1N) return;

  float best = pS[(size_t)row * NSPLIT];
  int   bi   = pI[(size_t)row * NSPLIT];
#pragma unroll
  for (int s = 1; s < NSPLIT; ++s) {
    float v = pS[(size_t)row * NSPLIT + s];
    int   i = pI[(size_t)row * NSPLIT + s];
    if (v < best || (v == best && i < bi)) { best = v; bi = i; }
  }
  float d2 = na[row] + best;
  d2 = d2 < 0.0f ? 0.0f : d2;
  outD[row] = __builtin_sqrtf(d2);
  outIdx[2 * row]     = row;
  outIdx[2 * row + 1] = bi;
}

// ---------------------------------------------------------------------------
extern "C" void kernel_launch(void* const* d_in, const int* in_sizes, int n_in,
                              void* d_out, int out_size, void* d_ws, size_t ws_size,
                              hipStream_t stream) {
  (void)in_sizes; (void)n_in; (void)out_size; (void)ws_size;

  const float* desc1 = (const float*)d_in[0];
  const float* desc2 = (const float*)d_in[1];

  // Workspace layout (~8.6 MB):
  //   aHi/aLo/bHi/bLo : 4 x 8192*128 bf16 (8 MB)
  //   na, nb          : 8192 f32 each
  //   pS, pI          : 8192*NSPLIT f32 / i32 partial winners
  __bf16* aHi = (__bf16*)d_ws;
  __bf16* aLo = aHi + (size_t)B1N * DD;
  __bf16* bHi = aLo + (size_t)B1N * DD;
  __bf16* bLo = bHi + (size_t)B2N * DD;
  float*  na  = (float*)(bLo + (size_t)B2N * DD);
  float*  nb  = na + B1N;
  float*  pS  = nb + B2N;
  int*    pI  = (int*)(pS + (size_t)B1N * NSPLIT);

  prep_kernel<<<B1N, DD, 0, stream>>>(desc1, aHi, aLo, na);
  prep_kernel<<<B2N, DD, 0, stream>>>(desc2, bHi, bLo, nb);

  dim3 grid(NSPLIT, B1N / MROWS_PER_WG);
  nn_wmma_kernel<<<grid, 256, 0, stream>>>(aHi, aLo, bHi, bLo, nb, pS, pI);

  float* outD   = (float*)d_out;
  int*   outIdx = (int*)(outD + B1N);
  finalize_kernel<<<(B1N + 255) / 256, 256, 0, stream>>>(pS, pI, na, outD, outIdx);
}
